// SpatialAttention_17540646437012
// MI455X (gfx1250) — compile-verified, matrix operand order checked by reading, not run on hardware
//
#include <hip/hip_runtime.h>
#include <hip/hip_bf16.h>
#include <cstdint>

// ---------------------------------------------------------------------------
// Types for CDNA5 WMMA (wave32): A/B = v16h (16x32 f16 fragment), C/D = v8f.
// ---------------------------------------------------------------------------
typedef __attribute__((ext_vector_type(16))) _Float16 v16h;
typedef __attribute__((ext_vector_type(8)))  _Float16 v8h_t;
typedef __attribute__((ext_vector_type(8)))  float    v8f;
typedef __attribute__((ext_vector_type(4)))  int      v4i_t;
typedef __attribute__((ext_vector_type(4)))  unsigned int u32x4;
typedef __attribute__((ext_vector_type(8)))  int      i32x8;
typedef __attribute__((ext_vector_type(4)))  int      i32x4;

#define TILE     128                 // block tile (M and N)
#define KSTEP    32                  // K per WMMA step
#define LDSS     40                  // halves per LDS row (32 data + 8 pad, 80B)
#define BUFH     (TILE * LDSS)       // halves per LDS tile buffer
#define NTHREADS 256                 // 8 wave32s

#define AS3 __attribute__((address_space(3)))
#define AS1 __attribute__((address_space(1)))

#if __has_builtin(__builtin_amdgcn_tensor_load_to_lds)
#define HAVE_TDM 1
#else
#define HAVE_TDM 0
#endif

union FragU { v16h v; v8h_t h[2]; };

// ---------------------------------------------------------------------------
// Async global->LDS 16B copy (GLOBAL_LOAD_ASYNC_TO_LDS_B128, ASYNCcnt).
// ---------------------------------------------------------------------------
__device__ __forceinline__ void async_b128(const _Float16* gsrc, _Float16* ldst) {
#if __has_builtin(__builtin_amdgcn_global_load_async_to_lds_b128)
    AS1 v4i_t* g = (AS1 v4i_t*)(uintptr_t)gsrc;           // global AS1
    AS3 v4i_t* l = (AS3 v4i_t*)(uint32_t)(uintptr_t)ldst; // LDS offset = addr[31:0]
    __builtin_amdgcn_global_load_async_to_lds_b128(g, l, 0, 0);
#else
    *(uint4*)ldst = *(const uint4*)gsrc;
#endif
}

__device__ __forceinline__ void async_wait0() {
#if __has_builtin(__builtin_amdgcn_s_wait_asynccnt)
    __builtin_amdgcn_s_wait_asynccnt(0);
#endif
}

__device__ __forceinline__ void tensor_wait0() {
#if __has_builtin(__builtin_amdgcn_s_wait_tensorcnt)
    __builtin_amdgcn_s_wait_tensorcnt(0);
#endif
}

// ---------------------------------------------------------------------------
// TDM: DMA one 128(row) x 32(col) f16 tile into LDS with padded rows.
// D# (CDNA5 ISA): group0 = {count=1, lds_addr, global_addr[56:0], type=2},
// group1: data_size=1(2B), pad_enable, pad_interval=16dw (code 3) -> pad after
// each 64B row, pad_amount=4dw (code 3) -> +16B => LDS row stride 80B = LDSS.
// tensor_dim1 = rows_valid gives HW zero-fill for OOB rows.
// Builtin here is the 6-arg (clang-23 / therock) form:
//   (u32x4 g0, i32x8 g1, i32x4 g2, i32x4 g3, i32x8 g4, i32 cpol)
// ---------------------------------------------------------------------------
__device__ __forceinline__ void tdm_stage(const _Float16* gtile, _Float16* ltile,
                                          int ld, int rows_valid) {
#if HAVE_TDM
    unsigned long long ga = (unsigned long long)(uintptr_t)gtile;
    unsigned int la = (unsigned int)(uintptr_t)ltile;
    u32x4 g0 = { 1u, la, (unsigned int)ga,
                 (unsigned int)((ga >> 32) & 0x01ffffffu) | 0x80000000u };
    unsigned int uld = (unsigned int)ld, urv = (unsigned int)rows_valid;
    i32x8 g1 = {
        (int)((1u << 16) | (1u << 20) | (3u << 22) | (3u << 25)), // dsz=2B,pad en,interval,amount
        (int)((uld & 0xffffu) << 16),                             // tensor_dim0[15:0] @ bits63:48
        (int)(((uld >> 16) & 0xffffu) | ((urv & 0xffffu) << 16)), // dim0[31:16], dim1[15:0]
        (int)(((urv >> 16) & 0xffffu) | (32u << 16)),             // dim1[31:16], tile_dim0=32
        128,                                                      // tile_dim1=128, tile_dim2=0
        ld,                                                       // dim0_stride[31:0]
        0, 0                                                      // stride hi, dim1_stride
    };
    i32x4 z4 = { 0, 0, 0, 0 };
    i32x8 z8 = { 0, 0, 0, 0, 0, 0, 0, 0 };
    __builtin_amdgcn_tensor_load_to_lds(g0, g1, z4, z4, z8, 0);
#else
    (void)gtile; (void)ltile; (void)ld; (void)rows_valid;
#endif
}

// Load one 16x32 f16 A/B fragment from an LDS tile stored [row][k], stride
// LDSS halves. Lanes 0-15: row lane&15, K halves {kb..kb+7, kb+16..kb+23},
// kb = (lane>=16) ? 8 : 0.
__device__ __forceinline__ v16h frag_load(const _Float16* sm, int row0, int lane) {
    int r  = row0 + (lane & 15);
    int kb = (lane >> 4) << 3;
    const _Float16* p = sm + r * LDSS + kb;
    FragU f;
    f.h[0] = *(const v8h_t*)(p);
    f.h[1] = *(const v8h_t*)(p + 16);
    return f.v;
}

// 8 WMMAs per K-step: wave tile 64(M) x 32(N) as 4x2 grid of 16x16.
__device__ __forceinline__ void gemm_core(const _Float16* As, const _Float16* Bs,
                                          v8f acc[4][2], int wm, int wn, int lane) {
    v16h a[4], b[2];
#pragma unroll
    for (int i = 0; i < 4; ++i) a[i] = frag_load(As, wm * 64 + i * 16, lane);
#pragma unroll
    for (int j = 0; j < 2; ++j) b[j] = frag_load(Bs, wn * 32 + j * 16, lane);
#pragma unroll
    for (int i = 0; i < 4; ++i)
#pragma unroll
        for (int j = 0; j < 2; ++j)
            acc[i][j] = __builtin_amdgcn_wmma_f32_16x16x32_f16(
                false, a[i], false, b[j], (short)0, acc[i][j], false, false);
}

// ---------------------------------------------------------------------------
// Double-buffered K-loop. A tile staged by the TDM (wave 0 issues one
// tensor_load_to_lds per stage; TENSORcnt). B tile staged by all waves with
// async b128 copies (ASYNCcnt); per-thread addresses hoisted out of the loop.
// ---------------------------------------------------------------------------
__device__ __forceinline__ void gemm_loop(const _Float16* __restrict__ A, long lda, int adim, int m0,
                                          const _Float16* __restrict__ B, long ldb, int bdim, int n0,
                                          long K, v8f acc[4][2],
                                          _Float16* As, _Float16* Bs,
                                          int tid, int wm, int wn, int lane) {
    const long nk = K / KSTEP;
    const int w = tid >> 5;

    // Per-thread async staging addresses for B (2 chunks: rows r0 and r0+64).
    int r0 = tid >> 2, kc = (tid & 3) << 3;
    int gb0 = n0 + r0;      gb0 = (gb0 < bdim) ? gb0 : bdim - 1;
    int gb1 = n0 + r0 + 64; gb1 = (gb1 < bdim) ? gb1 : bdim - 1;
    const _Float16* gB0 = B + (long)gb0 * ldb + kc;
    const _Float16* gB1 = B + (long)gb1 * ldb + kc;
    const int lofs = r0 * LDSS + kc;

#if HAVE_TDM
    const _Float16* gA = A + (long)m0 * lda;   // tile row start; +k*KSTEP per stage
    const int rowsA = adim - m0;
#else
    int ga0 = m0 + r0;      ga0 = (ga0 < adim) ? ga0 : adim - 1;
    int ga1 = m0 + r0 + 64; ga1 = (ga1 < adim) ? ga1 : adim - 1;
    const _Float16* gA0 = A + (long)ga0 * lda + kc;
    const _Float16* gA1 = A + (long)ga1 * lda + kc;
#endif

    auto stageA = [&](long k) {
        _Float16* a = As + (k & 1) * BUFH;
#if HAVE_TDM
        if (w == 0) tdm_stage(gA + k * KSTEP, a, (int)lda, rowsA);
#else
        async_b128(gA0 + k * KSTEP, a + lofs);
        async_b128(gA1 + k * KSTEP, a + lofs + 64 * LDSS);
#endif
    };
    auto stageB = [&](long k) {
        _Float16* bsm = Bs + (k & 1) * BUFH;
        async_b128(gB0 + k * KSTEP, bsm + lofs);
        async_b128(gB1 + k * KSTEP, bsm + lofs + 64 * LDSS);
    };

    stageA(0);
    stageB(0);
    async_wait0(); tensor_wait0();
    __syncthreads();
    for (long k = 0; k < nk; ++k) {
        if (k + 1 < nk) { stageA(k + 1); stageB(k + 1); }
        gemm_core(As + (k & 1) * BUFH, Bs + (k & 1) * BUFH, acc, wm, wn, lane);
        async_wait0(); tensor_wait0();
        __syncthreads();
    }
}

// ---------------------------------------------------------------------------
// Fold conv bias + BN into (W', b'): y = relu(W'x + b').
// ---------------------------------------------------------------------------
__global__ void fold_kernel(const float* __restrict__ W, const float* __restrict__ b,
                            const float* __restrict__ g, const float* __restrict__ be,
                            const float* __restrict__ mn, const float* __restrict__ vr,
                            _Float16* __restrict__ Wf, float* __restrict__ bf, int C) {
    int idx = blockIdx.x * blockDim.x + threadIdx.x;
    if (idx >= C * C) return;
    int o = idx / C;
    float sc = g[o] * rsqrtf(vr[o] + 1e-5f);
    Wf[idx] = (_Float16)(W[idx] * sc);
    if ((idx - o * C) == 0) bf[o] = b[o] * sc + be[o] - mn[o] * sc;
}

// ---------------------------------------------------------------------------
// Convert + transpose inputs once: Xt[b][n][c] = (f16) X[b][c][n].
// ---------------------------------------------------------------------------
__global__ void cvt_transpose_kernel(const float* __restrict__ X, _Float16* __restrict__ Xt,
                                     int C, int N, long total) {
    long idx = (long)blockIdx.x * blockDim.x + threadIdx.x;
    if (idx >= total) return;
    long cn = (long)C * N;
    long b = idx / cn, rem = idx - b * cn;
    long n = rem / C, c = rem - n * C;
    Xt[idx] = (_Float16)X[b * cn + c * N + n];
}

// ---------------------------------------------------------------------------
// CBR GEMM: Y = relu(W' @ X + b'), f16 out. transOut=1 -> Y[n][o] (d1,d2).
// ---------------------------------------------------------------------------
__global__ __launch_bounds__(NTHREADS)
void cbr_gemm_kernel(const _Float16* __restrict__ Wf, const float* __restrict__ bias,
                     const _Float16* __restrict__ Xt, _Float16* __restrict__ Y,
                     int C, int Nsp, int transOut) {
    __shared__ _Float16 As[2 * BUFH];
    __shared__ _Float16 Bs[2 * BUFH];
    int tid = threadIdx.x, lane = tid & 31, w = tid >> 5;
    int wm = w >> 2, wn = w & 3;
    int n0 = blockIdx.x * TILE, m0 = blockIdx.y * TILE;
    long bofs = (long)blockIdx.z * C * Nsp;
    Xt += bofs; Y += bofs;

    v8f acc[4][2] = {};
    gemm_loop(Wf, C, C, m0, Xt, C, Nsp, n0, C, acc, As, Bs, tid, wm, wn, lane);

#pragma unroll
    for (int i = 0; i < 4; ++i)
#pragma unroll
        for (int j = 0; j < 2; ++j) {
            int mb = m0 + wm * 64 + i * 16 + ((lane >> 4) << 3);
            int nc = n0 + wn * 32 + j * 16 + (lane & 15);
#pragma unroll
            for (int v = 0; v < 8; ++v) {
                int m = mb + v;
                if (m < C && nc < Nsp) {
                    float val = acc[i][j][v] + bias[m];
                    _Float16 o = (_Float16)(val > 0.f ? val : 0.f);
                    if (transOut) Y[(long)nc * C + m] = o;
                    else          Y[(long)m * Nsp + nc] = o;
                }
            }
        }
}

// ---------------------------------------------------------------------------
// Score GEMM (stored transposed -> softmax row-wise):
// St[m,n] = sum_c d2t[m,c] * d1t[n,c]. f32 out.
// ---------------------------------------------------------------------------
__global__ __launch_bounds__(NTHREADS)
void score_gemm_kernel(const _Float16* __restrict__ Qt, const _Float16* __restrict__ Kt,
                       float* __restrict__ St, int C, int Nsp) {
    __shared__ _Float16 As[2 * BUFH];
    __shared__ _Float16 Bs[2 * BUFH];
    int tid = threadIdx.x, lane = tid & 31, w = tid >> 5;
    int wm = w >> 2, wn = w & 3;
    int n0 = blockIdx.x * TILE, m0 = blockIdx.y * TILE;

    v8f acc[4][2] = {};
    gemm_loop(Qt, C, Nsp, m0, Kt, C, Nsp, n0, C, acc, As, Bs, tid, wm, wn, lane);

#pragma unroll
    for (int i = 0; i < 4; ++i)
#pragma unroll
        for (int j = 0; j < 2; ++j) {
            int mb = m0 + wm * 64 + i * 16 + ((lane >> 4) << 3);
            int nc = n0 + wn * 32 + j * 16 + (lane & 15);
#pragma unroll
            for (int v = 0; v < 8; ++v) {
                int m = mb + v;
                if (m < Nsp && nc < Nsp) St[(long)m * Nsp + nc] = acc[i][j][v];
            }
        }
}

// ---------------------------------------------------------------------------
// Row softmax: P[m,:] = softmax(St[m,:]) written as f16.
// ---------------------------------------------------------------------------
__global__ __launch_bounds__(NTHREADS)
void softmax_row_kernel(const float* __restrict__ St, _Float16* __restrict__ Ph, int N) {
    __shared__ float red[NTHREADS];
    int m = blockIdx.x, tid = threadIdx.x;
    const float* row = St + (long)m * N;

    float mx = -3.4e38f;
    for (int i = tid; i < N; i += NTHREADS) mx = fmaxf(mx, row[i]);
    red[tid] = mx; __syncthreads();
    for (int s = NTHREADS / 2; s > 0; s >>= 1) {
        if (tid < s) red[tid] = fmaxf(red[tid], red[tid + s]);
        __syncthreads();
    }
    mx = red[0]; __syncthreads();

    float sum = 0.f;
    for (int i = tid; i < N; i += NTHREADS) sum += __expf(row[i] - mx);
    red[tid] = sum; __syncthreads();
    for (int s = NTHREADS / 2; s > 0; s >>= 1) {
        if (tid < s) red[tid] += red[tid + s];
        __syncthreads();
    }
    float inv = 1.f / red[0];
    for (int i = tid; i < N; i += NTHREADS)
        Ph[(long)m * N + i] = (_Float16)(__expf(row[i] - mx) * inv);
}

// ---------------------------------------------------------------------------
// AV GEMM + residual: Out[c,m] = sum_n R[c,n]*P[m,n] + R[c,m].
// ---------------------------------------------------------------------------
__global__ __launch_bounds__(NTHREADS)
void av_gemm_kernel(const _Float16* __restrict__ Rh, const _Float16* __restrict__ Ph,
                    float* __restrict__ Out, int C, int Nsp) {
    __shared__ _Float16 As[2 * BUFH];
    __shared__ _Float16 Bs[2 * BUFH];
    int tid = threadIdx.x, lane = tid & 31, w = tid >> 5;
    int wm = w >> 2, wn = w & 3;
    int n0 = blockIdx.x * TILE, m0 = blockIdx.y * TILE;

    v8f acc[4][2] = {};
    gemm_loop(Rh, Nsp, C, m0, Ph, Nsp, Nsp, n0, Nsp, acc, As, Bs, tid, wm, wn, lane);

#pragma unroll
    for (int i = 0; i < 4; ++i)
#pragma unroll
        for (int j = 0; j < 2; ++j) {
            int mb = m0 + wm * 64 + i * 16 + ((lane >> 4) << 3);
            int nc = n0 + wn * 32 + j * 16 + (lane & 15);
#pragma unroll
            for (int v = 0; v < 8; ++v) {
                int m = mb + v;
                if (m < C && nc < Nsp)
                    Out[(long)m * Nsp + nc] = acc[i][j][v] + (float)Rh[(long)m * Nsp + nc];
            }
        }
}

// ---------------------------------------------------------------------------
// Host-side orchestration.
// ---------------------------------------------------------------------------
static const int  DIMC[4]   = {256, 512, 1024, 2048};
static const int  DIMN[4]   = {4096, 1024, 256, 64};
static const long OUTOFS[4] = {0, 8388608, 12582912, 14680064};

extern "C" void kernel_launch(void* const* d_in, const int* in_sizes, int n_in,
                              void* d_out, int out_size, void* d_ws, size_t ws_size,
                              hipStream_t stream) {
    (void)in_sizes; (void)n_in; (void)out_size; (void)ws_size;
    const int B = 8;

    for (int i = 0; i < 4; ++i) {
        const int C = DIMC[i], N = DIMN[i];
        const float* Xrgb = (const float*)d_in[2 * i];
        const float* Xd   = (const float*)d_in[2 * i + 1];
        const long cn = (long)C * N, total = (long)B * cn;

        // Carve workspace (reused across scales; stream-ordered so safe).
        char* p = (char*)d_ws;
        auto carve = [&p](size_t bytes) -> char* {
            char* r = p; p += (bytes + 255) & ~(size_t)255; return r;
        };
        _Float16* Wf[3]; float* bf[3];
        for (int j = 0; j < 3; ++j) {
            Wf[j] = (_Float16*)carve((size_t)C * C * 2);
            bf[j] = (float*)carve((size_t)C * 4);
        }
        _Float16* Xtd = (_Float16*)carve((size_t)total * 2);  // (f16) d^T   [b][n][c]
        _Float16* Xtr = (_Float16*)carve((size_t)total * 2);  // (f16) rgb^T [b][n][c]
        _Float16* d1t = (_Float16*)carve((size_t)total * 2);  // d1 transposed [b][n][c]
        _Float16* d2t = (_Float16*)carve((size_t)total * 2);  // d2 transposed [b][n][c]
        _Float16* rh  = (_Float16*)carve((size_t)total * 2);  // r normal      [b][c][n]
        float*    St  = (float*)carve((size_t)N * N * 4);
        _Float16* Ph  = (_Float16*)carve((size_t)N * N * 2);

        // Fold BN into weights/bias for the 3 CBR branches (d1, d2, rgb).
        for (int j = 0; j < 3; ++j) {
            const int base = 8 + i * 18 + j * 6;
            fold_kernel<<<(C * C + 255) / 256, 256, 0, stream>>>(
                (const float*)d_in[base + 0], (const float*)d_in[base + 1],
                (const float*)d_in[base + 2], (const float*)d_in[base + 3],
                (const float*)d_in[base + 4], (const float*)d_in[base + 5],
                Wf[j], bf[j], C);
        }

        // Convert + transpose the two input tensors to f16 [b][n][c].
        int cb = (int)((total + 255) / 256);
        cvt_transpose_kernel<<<cb, 256, 0, stream>>>(Xd,   Xtd, C, N, total);
        cvt_transpose_kernel<<<cb, 256, 0, stream>>>(Xrgb, Xtr, C, N, total);

        // Three CBR GEMMs. d1/d2 written transposed.
        dim3 gc((N + TILE - 1) / TILE, (C + TILE - 1) / TILE, B);
        cbr_gemm_kernel<<<gc, NTHREADS, 0, stream>>>(Wf[0], bf[0], Xtd, d1t, C, N, 1);
        cbr_gemm_kernel<<<gc, NTHREADS, 0, stream>>>(Wf[1], bf[1], Xtd, d2t, C, N, 1);
        cbr_gemm_kernel<<<gc, NTHREADS, 0, stream>>>(Wf[2], bf[2], Xtr, rh,  C, N, 0);

        // Attention per batch: scores -> softmax -> AV + residual.
        float* outScale = (float*)d_out + OUTOFS[i];
        dim3 gs((N + TILE - 1) / TILE, (N + TILE - 1) / TILE);
        dim3 ga((N + TILE - 1) / TILE, (C + TILE - 1) / TILE);
        for (int b = 0; b < B; ++b) {
            long fo = b * cn;
            score_gemm_kernel<<<gs, NTHREADS, 0, stream>>>(d2t + fo, d1t + fo, St, C, N);
            softmax_row_kernel<<<N, NTHREADS, 0, stream>>>(St, Ph, N);
            av_gemm_kernel<<<ga, NTHREADS, 0, stream>>>(rh + fo, Ph, outScale + fo, C, N);
        }
    }
}
